// BERT_65343632441842
// MI455X (gfx1250) — matrix operand order repaired and checked by reading, not compiled, not run on hardware
//
#include <hip/hip_runtime.h>
#include <hip/hip_bf16.h>
#include <stdint.h>

// ---------------- constants (match reference) ----------------
#define BB 16
#define TT 512
#define DD 768
#define HH 12
#define HS 64
#define VV 32000
#define NMSK 20

typedef __attribute__((ext_vector_type(16))) __bf16 v16bf;
typedef __attribute__((ext_vector_type(8)))  float  v8f;

union FragBF { v16bf v; unsigned u[8]; };

static __device__ __forceinline__ unsigned short f2bf(float f) {
  unsigned u = __float_as_uint(f);
  u += 0x7fffu + ((u >> 16) & 1u);     // round-to-nearest-even
  return (unsigned short)(u >> 16);
}

// A fragment (16x32 bf16): lane L -> M = L%16 ; VGPR i packs K pair
// K = k0 + (i<4?0:16) + (L>=16?8:0) + 2*(i&3). Rows stored K-major, stride ldk.
static __device__ __forceinline__ v16bf load_A(const unsigned short* base, int ldk,
                                               int m0, int k0, int lane) {
  FragBF f;
  const unsigned* p = (const unsigned*)base;
  size_t row = (size_t)(m0 + (lane & 15)) * ldk;
  int kb = k0 + ((lane & 16) ? 8 : 0);
#pragma unroll
  for (int i = 0; i < 8; ++i) {
    int kk = kb + ((i < 4) ? 0 : 16) + 2 * (i & 3);
    f.u[i] = p[(row + kk) >> 1];
  }
  return f.v;
}

// B fragment (32x16 bf16): lane L -> N = L%16 ; K split by lane group;
// VGPR i packs K = k0 + (i<4?0:16) + (L>=16?8:0) + 2*(i&3). B stored N-major rows, K contiguous (stride ldk).
static __device__ __forceinline__ v16bf load_B(const unsigned short* base, int ldk,
                                               int n0, int k0, int lane) {
  FragBF f;
  const unsigned* p = (const unsigned*)base;
  size_t row = (size_t)(n0 + (lane & 15)) * ldk;
  int kb = k0 + ((lane & 16) ? 8 : 0);
#pragma unroll
  for (int i = 0; i < 8; ++i) f.u[i] = p[(row + kb + ((i < 4) ? 0 : 16) + 2 * (i & 3)) >> 1];
  return f.v;
}

static __device__ __forceinline__ v8f wmma_bf16(v16bf a, v16bf b, v8f c) {
  return __builtin_amdgcn_wmma_f32_16x16x32_bf16(false, a, false, b, (short)0, c, false, false);
}

// CDNA5 async copy global->LDS (b128 granules), completion via ASYNCcnt.
// Low 32 bits of a flat LDS pointer are the LDS byte offset (aperture rule),
// so they form the VDST operand directly. Caller must __syncthreads() after.
static __device__ __forceinline__ void async_copy_g2l(const void* gsrc, void* lds,
                                                      int bytes, int tid, int nthr) {
  unsigned lbase = (unsigned)(size_t)lds;
  unsigned long long g = (unsigned long long)(size_t)gsrc;
  for (int c = tid * 16; c < bytes; c += nthr * 16) {
    unsigned laddr = lbase + (unsigned)c;
    unsigned long long gaddr = g + (unsigned long long)c;
    asm volatile("global_load_async_to_lds_b128 %0, %1, off"
                 :: "v"(laddr), "v"(gaddr) : "memory");
  }
  asm volatile("s_wait_asynccnt 0x0" ::: "memory");
}

// ---------------- 1) f32 [M,R,C] -> bf16 [M,C,R] tiled transpose+convert ----------------
__global__ void k_transpose_cvt(const float* __restrict__ in, unsigned short* __restrict__ out,
                                int R, int C) {
  __shared__ float tile[32][33];
  int m = blockIdx.z;
  const float* inp = in + (size_t)m * R * C;
  unsigned short* outp = out + (size_t)m * R * C;
  int c0 = blockIdx.x * 32, r0 = blockIdx.y * 32;
  int tx = threadIdx.x, ty = threadIdx.y;
#pragma unroll
  for (int i = 0; i < 4; ++i) {
    int r = r0 + ty + i * 8, c = c0 + tx;
    if (r < R && c < C) tile[ty + i * 8][tx] = inp[(size_t)r * C + c];
  }
  __syncthreads();
#pragma unroll
  for (int i = 0; i < 4; ++i) {
    int c = c0 + ty + i * 8, r = r0 + tx;
    if (r < R && c < C) outp[(size_t)c * R + r] = f2bf(tile[tx][ty + i * 8]);
  }
}

// ---------------- 2) embedding + LayerNorm -> x bf16 [B,T,D] ----------------
__global__ void k_embed_ln(const int* __restrict__ seq, const int* __restrict__ seg,
                           const float* __restrict__ tok, const float* __restrict__ sge,
                           const float* __restrict__ pos, const float* __restrict__ g,
                           const float* __restrict__ bta, unsigned short* __restrict__ xbf) {
  int bt = blockIdx.x;                 // b*T + t
  int t = bt % TT;
  int tid = threadIdx.x;
  int sq = seq[bt], sg = seg[bt];
  float e[3]; float s = 0.f, q = 0.f;
#pragma unroll
  for (int j = 0; j < 3; ++j) {
    int d = tid + j * 256;
    float v = tok[(size_t)sq * DD + d] + sge[(size_t)sg * DD + d] + pos[(size_t)t * DD + d];
    e[j] = v; s += v; q += v * v;
  }
  __shared__ float rs_[256], rq_[256];
  rs_[tid] = s; rq_[tid] = q; __syncthreads();
  for (int o = 128; o > 0; o >>= 1) {
    if (tid < o) { rs_[tid] += rs_[tid + o]; rq_[tid] += rq_[tid + o]; }
    __syncthreads();
  }
  float mean = rs_[0] * (1.f / DD);
  float var  = rq_[0] * (1.f / DD) - mean * mean;
  float inv  = rsqrtf(var + 1e-5f);
#pragma unroll
  for (int j = 0; j < 3; ++j) {
    int d = tid + j * 256;
    xbf[(size_t)bt * DD + d] = f2bf((e[j] - mean) * inv * g[d] + bta[d]);
  }
}

// ---------------- 3) QKV projection: x[b] (TxD) @ W[h] (DxHS) via WMMA ----------------
// A-tile (16x768 bf16, 24KB, contiguous) staged once per block via async LDS copy,
// shared by 4 waves x 24 k-steps (cuts A global traffic 4x).
__global__ void k_qkv(const unsigned short* __restrict__ xbf,
                      const unsigned short* __restrict__ wqt, const unsigned short* __restrict__ wkt,
                      const unsigned short* __restrict__ wvt,
                      unsigned short* __restrict__ qo, unsigned short* __restrict__ ko,
                      unsigned short* __restrict__ vto) {
  __shared__ __align__(16) unsigned short sX[16 * DD];
  int bh = blockIdx.y, b = bh / HH, h = bh % HH;
  int t0 = blockIdx.x * 16;
  int wid = threadIdx.x >> 5, lane = threadIdx.x & 31;
  int s0 = wid * 16;                                   // 4 waves cover HS=64
  const unsigned short* xb  = xbf + ((size_t)b * TT + t0) * DD;
  const unsigned short* wqh = wqt + (size_t)h * HS * DD;
  const unsigned short* wkh = wkt + (size_t)h * HS * DD;
  const unsigned short* wvh = wvt + (size_t)h * HS * DD;

  async_copy_g2l(xb, sX, 16 * DD * 2, threadIdx.x, 128);
  __syncthreads();

  v8f aq = {}, ak = {}, av = {};
  for (int kc = 0; kc < DD / 32; ++kc) {
    int k0 = kc * 32;
    v16bf a = load_A(sX, DD, 0, k0, lane);             // A reused for 3 WMMAs
    aq = wmma_bf16(a, load_B(wqh, DD, s0, k0, lane), aq);
    ak = wmma_bf16(a, load_B(wkh, DD, s0, k0, lane), ak);
    av = wmma_bf16(a, load_B(wvh, DD, s0, k0, lane), av);
  }
  int n  = s0 + (lane & 15);
  int mh = (lane & 16) ? 8 : 0;
  size_t base = (size_t)bh * TT * HS;
#pragma unroll
  for (int r = 0; r < 8; ++r) {
    int m = t0 + r + mh;
    qo[base + (size_t)m * HS + n] = f2bf(aq[r]);
    ko[base + (size_t)m * HS + n] = f2bf(ak[r]);
    vto[(size_t)bh * HS * TT + (size_t)n * TT + m] = f2bf(av[r]);   // V transposed [HS,T]
  }
}

// ---------------- 4) attention for one (b,h,16-query tile); scores+softmax in LDS ----------------
__global__ void k_attn(const unsigned short* __restrict__ q, const unsigned short* __restrict__ k,
                       const unsigned short* __restrict__ vt, const int* __restrict__ amask,
                       float* __restrict__ ctx) {
  __shared__ float          sS[16][TT];    // 32 KB scores / exp values
  __shared__ unsigned short sP[16][TT];    // 16 KB bf16 probabilities
  __shared__ __align__(16) unsigned short sQ[16 * HS];  // 2 KB Q tile
  __shared__ float sRM[16][8], sRS[16][8];
  int bh = blockIdx.y, b = bh / HH, h = bh % HH;
  int t0 = blockIdx.x * 16;
  int wid = threadIdx.x >> 5, lane = threadIdx.x & 31;
  const unsigned short* qb = q  + ((size_t)bh * TT + t0) * HS;
  const unsigned short* kb = k  + (size_t)bh * TT * HS;
  const unsigned short* vb = vt + (size_t)bh * HS * TT;

  async_copy_g2l(qb, sQ, 16 * HS * 2, threadIdx.x, 128);
  __syncthreads();

  // scores S = Q K^T * HS^-0.5 ; reference masks over the QUERY dim with -1e-11
  for (int jt = 0; jt < 8; ++jt) {
    int j0 = wid * 128 + jt * 16;
    v8f acc = {};
#pragma unroll
    for (int kc = 0; kc < 2; ++kc) {
      v16bf a   = load_A(sQ, HS, 0, kc * 32, lane);
      v16bf bfr = load_B(kb, HS, j0, kc * 32, lane);   // B(kdim,n) = K[j0+n, kdim]
      acc = wmma_bf16(a, bfr, acc);
    }
    int n = lane & 15, mh = (lane & 16) ? 8 : 0;
#pragma unroll
    for (int r = 0; r < 8; ++r) {
      int m = r + mh;
      float vsc = acc[r] * 0.125f;                     // HS^-0.5 = 1/8
      if (amask[b * TT + t0 + m] == 0) vsc = -1e-11f;
      sS[m][j0 + n] = vsc;
    }
  }
  __syncthreads();

  // softmax over 512 keys: 8 threads per row
  int row = threadIdx.x >> 3, gg = threadIdx.x & 7;
  float mx = -3.4e38f;
  for (int c = gg * 64; c < gg * 64 + 64; ++c) mx = fmaxf(mx, sS[row][c]);
  sRM[row][gg] = mx; __syncthreads();
  float rmx = sRM[row][0];
#pragma unroll
  for (int i = 1; i < 8; ++i) rmx = fmaxf(rmx, sRM[row][i]);
  float ssum = 0.f;
  for (int c = gg * 64; c < gg * 64 + 64; ++c) {
    float ev = __expf(sS[row][c] - rmx); sS[row][c] = ev; ssum += ev;
  }
  sRS[row][gg] = ssum; __syncthreads();
  float tot = 0.f;
#pragma unroll
  for (int i = 0; i < 8; ++i) tot += sRS[row][i];
  float invs = 1.f / tot;
  for (int c = gg * 64; c < gg * 64 + 64; ++c) sP[row][c] = f2bf(sS[row][c] * invs);
  __syncthreads();

  // out = P (16x512) @ V (512xHS); A fragments straight from LDS
  int n0 = wid * 16;
  v8f acc = {};
  const unsigned* spu = (const unsigned*)&sP[0][0];
  int m = lane & 15;
  int kbse = (lane & 16) ? 8 : 0;
  for (int kc = 0; kc < 16; ++kc) {
    FragBF fa;
#pragma unroll
    for (int i = 0; i < 8; ++i) {
      int kk = kc * 32 + kbse + ((i < 4) ? 0 : 16) + 2 * (i & 3);
      fa.u[i] = spu[(m * TT + kk) >> 1];
    }
    v16bf bfr = load_B(vb, TT, n0, kc * 32, lane);     // B(key,n) = Vt[n0+n, key]
    acc = wmma_bf16(fa.v, bfr, acc);
  }
  int mh = (lane & 16) ? 8 : 0;
#pragma unroll
  for (int r = 0; r < 8; ++r) {
    int mm = r + mh;
    ctx[((size_t)b * TT + t0 + mm) * DD + h * HS + n0 + (lane & 15)] = acc[r];
  }
}

// ---------------- 5) gather masked positions -> bf16 [B*NMSK, D] ----------------
__global__ void k_gather(const float* __restrict__ ctx, const int* __restrict__ mpos,
                         unsigned short* __restrict__ xm) {
  int blk = blockIdx.x;                // b*NMSK + i
  int b = blk / NMSK;
  int t = mpos[blk];
  int tid = threadIdx.x;
#pragma unroll
  for (int j = 0; j < 3; ++j) {
    int d = tid + j * 256;
    xm[(size_t)blk * DD + d] = f2bf(ctx[((size_t)b * TT + t) * DD + d]);
  }
}

// ---------------- 6) LM head: [320,768] @ [768,32000] + bias ----------------
// A-tile staged via async LDS copy; B streamed from global with prefetch.
__global__ void k_lm_head(const unsigned short* __restrict__ xm,
                          const unsigned short* __restrict__ wlmt,
                          const float* __restrict__ blm, float* __restrict__ out) {
  __shared__ __align__(16) unsigned short sA[16 * DD];
  int wid = threadIdx.x >> 5, lane = threadIdx.x & 31;
  int n0 = blockIdx.x * 64 + wid * 16;
  int m0 = blockIdx.y * 16;

  async_copy_g2l(xm + (size_t)m0 * DD, sA, 16 * DD * 2, threadIdx.x, 128);
  __syncthreads();

  v8f acc = {};
  for (int kc = 0; kc < DD / 32; ++kc) {
    int k0 = kc * 32;
    if (k0 + 64 < DD)                   // global_prefetch_b8 on the weight stream
      __builtin_prefetch((const void*)(wlmt + (size_t)(n0 + (lane & 15)) * DD + k0 + 64), 0, 1);
    v16bf a   = load_A(sA,   DD, 0,  k0, lane);
    v16bf bfr = load_B(wlmt, DD, n0, k0, lane);
    acc = wmma_bf16(a, bfr, acc);
  }
  int n = n0 + (lane & 15);
  int mh = (lane & 16) ? 8 : 0;
  float bias = blm[n];
#pragma unroll
  for (int r = 0; r < 8; ++r)
    out[(size_t)(m0 + r + mh) * VV + n] = acc[r] + bias;
}

// ---------------- 7) classifier head ----------------
__global__ void k_cls(const float* __restrict__ ctx, const float* __restrict__ wc,
                      const float* __restrict__ bc, float* __restrict__ out2) {
  int b = blockIdx.x, tid = threadIdx.x;
  float a0 = 0.f, a1 = 0.f;
  for (int d = tid; d < DD; d += 128) {
    float xv = ctx[(size_t)b * TT * DD + d];     // token 0
    a0 += xv * wc[d * 2 + 0];
    a1 += xv * wc[d * 2 + 1];
  }
  __shared__ float r0[128], r1[128];
  r0[tid] = a0; r1[tid] = a1; __syncthreads();
  for (int o = 64; o > 0; o >>= 1) {
    if (tid < o) { r0[tid] += r0[tid + o]; r1[tid] += r1[tid + o]; }
    __syncthreads();
  }
  if (tid == 0) {
    out2[b * 2 + 0] = r0[0] + bc[0];
    out2[b * 2 + 1] = r1[0] + bc[1];
  }
}

// ---------------- launcher ----------------
extern "C" void kernel_launch(void* const* d_in, const int* in_sizes, int n_in,
                              void* d_out, int out_size, void* d_ws, size_t ws_size,
                              hipStream_t stream) {
  (void)in_sizes; (void)n_in; (void)out_size; (void)ws_size;
  const int*   seq   = (const int*)d_in[0];
  const int*   seg   = (const int*)d_in[1];
  const int*   amask = (const int*)d_in[2];
  const int*   mpos  = (const int*)d_in[3];
  const float* tok   = (const float*)d_in[4];
  const float* sge   = (const float*)d_in[5];
  const float* pos   = (const float*)d_in[6];
  const float* lng   = (const float*)d_in[7];
  const float* lnb   = (const float*)d_in[8];
  const float* wq    = (const float*)d_in[9];
  const float* wk    = (const float*)d_in[10];
  const float* wv    = (const float*)d_in[11];
  const float* wlm   = (const float*)d_in[12];
  const float* blm   = (const float*)d_in[13];
  const float* wcls  = (const float*)d_in[14];
  const float* bcls  = (const float*)d_in[15];

  char* ws = (char*)d_ws;
  size_t off = 0;
  auto alloc = [&](size_t bytes) {
    void* p = ws + off;
    off += (bytes + 255) & ~(size_t)255;
    return p;
  };
  unsigned short* xbf  = (unsigned short*)alloc((size_t)BB * TT * DD * 2);
  unsigned short* wqt  = (unsigned short*)alloc((size_t)HH * HS * DD * 2);
  unsigned short* wkt  = (unsigned short*)alloc((size_t)HH * HS * DD * 2);
  unsigned short* wvt  = (unsigned short*)alloc((size_t)HH * HS * DD * 2);
  unsigned short* qb   = (unsigned short*)alloc((size_t)BB * HH * TT * HS * 2);
  unsigned short* kbuf = (unsigned short*)alloc((size_t)BB * HH * TT * HS * 2);
  unsigned short* vtb  = (unsigned short*)alloc((size_t)BB * HH * HS * TT * 2);
  float*          ctx  = (float*)         alloc((size_t)BB * TT * DD * 4);
  unsigned short* xm   = (unsigned short*)alloc((size_t)BB * NMSK * DD * 2);
  unsigned short* wlmt = (unsigned short*)alloc((size_t)VV * DD * 2);

  dim3 tb(32, 8);
  // weight convert/transpose: Wq/Wk/Wv [H,D,HS] -> [H,HS,D] bf16 ; W_lm [D,V] -> [V,D] bf16
  k_transpose_cvt<<<dim3(HS / 32, DD / 32, HH), tb, 0, stream>>>(wq, wqt, DD, HS);
  k_transpose_cvt<<<dim3(HS / 32, DD / 32, HH), tb, 0, stream>>>(wk, wkt, DD, HS);
  k_transpose_cvt<<<dim3(HS / 32, DD / 32, HH), tb, 0, stream>>>(wv, wvt, DD, HS);
  k_transpose_cvt<<<dim3(VV / 32, DD / 32, 1),  tb, 0, stream>>>(wlm, wlmt, DD, VV);

  k_embed_ln<<<BB * TT, 256, 0, stream>>>(seq, seg, tok, sge, pos, lng, lnb, xbf);
  k_qkv<<<dim3(TT / 16, BB * HH), 128, 0, stream>>>(xbf, wqt, wkt, wvt, qb, kbuf, vtb);
  k_attn<<<dim3(TT / 16, BB * HH), 128, 0, stream>>>(qb, kbuf, vtb, amask, ctx);
  k_gather<<<BB * NMSK, 256, 0, stream>>>(ctx, mpos, xm);

  float* out = (float*)d_out;
  k_lm_head<<<dim3(VV / 64, (BB * NMSK) / 16), 128, 0, stream>>>(xm, wlmt, blm, out);
  k_cls<<<BB, 128, 0, stream>>>(ctx, wcls, bcls, out + (size_t)BB * NMSK * VV);
}
